// FullAttention_46969762349726
// MI455X (gfx1250) — compile-verified, matrix operand order checked by reading, not run on hardware
//
#include <hip/hip_runtime.h>
#include <math.h>

// Problem constants (from reference)
#define N_    8
#define L_    1024
#define S_    1024
#define H_    16
#define E_    64
#define MAXP_ 128
#define NJ_   257          // distance indices 0..256

// Tiling
#define BM    64           // query rows per block (4 waves x 16)
#define BKV   64           // keys per S tile
#define NWAVE 4
#define PADH  72           // f16 LDS row stride (halfs) for 64-wide tiles
#define QDPH  280          // f16 LDS row stride for Qd (272 cols + pad)
#define KDPH  152          // f16 LDS row stride for Kd band (144 cols + pad)
#define DROWS 272          // dist_emb rows padded to 17*16

typedef __attribute__((ext_vector_type(16))) _Float16 v16h;
typedef __attribute__((ext_vector_type(8)))  _Float16 v8h;
typedef __attribute__((ext_vector_type(4)))  _Float16 v4h;
typedef __attribute__((ext_vector_type(8)))  float    v8f;

// LDS layout (bytes); all tile bases 16B-aligned
#define OFF_SQ   0
#define SZ_SQ    (BM * PADH * 2)            // 9216
#define OFF_SK   (OFF_SQ + SZ_SQ)
#define SZ_SK    (BKV * PADH * 2)           // 9216
#define OFF_SVT  (OFF_SK + SZ_SK)
#define SZ_SVT   (E_ * PADH * 2)            // 9216
#define OFF_SP   (OFF_SVT + SZ_SVT)
#define SZ_SP    (NWAVE * 16 * PADH * 2)    // 9216
#define OFF_SD   (OFF_SP + SZ_SP)
#define SZ_SD    (DROWS * PADH * 2)         // 39168
#define OFF_SM   (OFF_SD + SZ_SD)
#define SZ_SM    (BM * BKV * 4)             // 16384 (mask tile, f32)
#define OFF_SQD  (OFF_SM + SZ_SM)
#define SZ_SQD   (BM * QDPH * 2)            // 35840 (f16)
#define OFF_SKD  (OFF_SQD + SZ_SQD)
#define SZ_SKD   (BKV * KDPH * 2)           // 19456 (f16)
#define SMEM_BYTES (OFF_SKD + SZ_SKD)       // 147712 -> 2 blocks / WGP

// Build a 16-half WMMA A/B fragment from padded row-major f16 LDS.
// Per-lane layout (CDNA5 16-bit A, mirrored for B):
//   row = row0 + (lane & 15); lanes 0-15 cover K-halves {0..7,16..23},
//   lanes 16-31 cover {8..15,24..31} (plus kstep*32).
__device__ __forceinline__ v16h load_frag(const _Float16* __restrict__ base,
                                          int row0, int kstep, int lane) {
    const int row = row0 + (lane & 15);
    const int kh  = (lane >> 4) & 1;
    const _Float16* p = base + row * PADH + kstep * 32 + kh * 8;
    v8h lo = *(const v8h*)(p);        // ds_load_b128: K = kbase+0..7
    v8h hi = *(const v8h*)(p + 16);   // ds_load_b128: K = kbase+16..23
    return __builtin_shufflevector(lo, hi,
        0, 1, 2, 3, 4, 5, 6, 7, 8, 9, 10, 11, 12, 13, 14, 15);
}

__device__ __forceinline__ v8f wmma16(v16h a, v16h b, v8f c) {
    return __builtin_amdgcn_wmma_f32_16x16x32_f16(
        false, a, false, b, (short)0, c, false, false);
}

__device__ __forceinline__ v4h cvt4(float4 t4) {
    v4h h;
    h[0] = (_Float16)t4.x; h[1] = (_Float16)t4.y;
    h[2] = (_Float16)t4.z; h[3] = (_Float16)t4.w;
    return h;
}

// gfx1250 async bulk copy global -> LDS (tracked with ASYNCcnt).
// lds_off is the wave-relative LDS byte address (low 32 bits of generic ptr).
__device__ __forceinline__ void async_load_b128(unsigned lds_off, const void* g) {
    asm volatile("global_load_async_to_lds_b128 %0, %1, off"
                 :: "v"(lds_off), "v"(g) : "memory");
}
__device__ __forceinline__ void wait_async0() {
    asm volatile("s_wait_asynccnt 0x0" ::: "memory");
}

__global__ __launch_bounds__(128)
void FullAttention_relpos_kernel(const float* __restrict__ q,
                                 const float* __restrict__ kk,
                                 const float* __restrict__ vv,
                                 const float* __restrict__ mask,
                                 const float* __restrict__ demb,
                                 float* __restrict__ out) {
    extern __shared__ char smem[];
    _Float16* sQ   = (_Float16*)(smem + OFF_SQ);   // [64][72] f16, Q rows
    _Float16* sK   = (_Float16*)(smem + OFF_SK);   // [64][72] f16, K rows
    _Float16* sVt  = (_Float16*)(smem + OFF_SVT);  // [64][72] f16, V^T: [e][key]
    _Float16* sP   = (_Float16*)(smem + OFF_SP);   // per-wave [16][72] f16 probs
    _Float16* sD   = (_Float16*)(smem + OFF_SD);   // [272][72] f16, dist_emb
    float*    sM   = (float*)(smem + OFF_SM);      // [64][64] f32 mask tile
    _Float16* sQd  = (_Float16*)(smem + OFF_SQD);  // [64][280] f16
    _Float16* sKd  = (_Float16*)(smem + OFF_SKD);  // [64][152] f16 (band)

    const int tid  = threadIdx.x;
    const int lane = tid & 31;
    const int wave = tid >> 5;
    const int hi8  = (lane >> 4) << 3;    // 0 or 8: C-layout row offset
    const int nj   = lane & 15;           // C-layout column within 16-tile

    const int l0 = blockIdx.x * BM;
    const int nh = blockIdx.y;
    const int n  = nh / H_;
    const int h  = nh % H_;

    const float temp = 0.125f;            // 1/sqrt(E)

    const float* qbase = q  + ((size_t)(n * L_ + l0) * H_ + h) * E_;
    const float* kbase = kk + ((size_t)(n * S_) * H_ + h) * E_;
    const float* vbase = vv + ((size_t)(n * S_) * H_ + h) * E_;

    const unsigned smask_lds = (unsigned)(size_t)(void*)sM;

    // ---- stage dist_emb (257 x 64 f32 -> f16) ----
    for (int c = tid; c < NJ_ * 16; c += 128) {
        const int j = c >> 4, e4 = c & 15;
        float4 t4 = *(const float4*)(demb + j * E_ + e4 * 4);
        *(v4h*)(sD + j * PADH + e4 * 4) = cvt4(t4);
    }
    // ---- stage Q tile (64 x 64) ----
    for (int c = tid; c < BM * 16; c += 128) {
        const int r = c >> 4, e4 = c & 15;
        float4 t4 = *(const float4*)(qbase + (size_t)r * (H_ * E_) + e4 * 4);
        *(v4h*)(sQ + r * PADH + e4 * 4) = cvt4(t4);
    }
    __syncthreads();

    // A fragments for this wave's 16 query rows (reused all kernel long)
    const v16h aq0 = load_frag(sQ, wave * 16, 0, lane);
    const v16h aq1 = load_frag(sQ, wave * 16, 1, lane);

    const v8f vz = {0.f, 0.f, 0.f, 0.f, 0.f, 0.f, 0.f, 0.f};

    // ---- Qd = Q_tile x demb^T over all 257 (padded 272) columns ----
    for (int t = 0; t < 17; ++t) {
        v16h b0 = load_frag(sD, t * 16, 0, lane);
        v16h b1 = load_frag(sD, t * 16, 1, lane);
        v8f c = vz;
        c = wmma16(aq0, b0, c);
        c = wmma16(aq1, b1, c);
#pragma unroll
        for (int r = 0; r < 8; ++r)
            sQd[(wave * 16 + r + hi8) * QDPH + t * 16 + nj] = (_Float16)c[r];
    }

    // Online-softmax state + output accumulators (16 rows x 64 cols per wave)
    float m_run[8], l_run[8];
    v8f oacc[4];
#pragma unroll
    for (int r = 0; r < 8; ++r) { m_run[r] = -INFINITY; l_run[r] = 0.f; }
#pragma unroll
    for (int t = 0; t < 4; ++t) oacc[t] = vz;

    __syncthreads();

    for (int st = 0; st < S_ / BKV; ++st) {
        const int s0 = st * BKV;
        __syncthreads();   // protect sK/sVt/sM from previous iteration's readers

        // async bulk copy of mask tile (f32, no conversion) into LDS
        for (int c = tid; c < BM * 16; c += 128) {
            const int r = c >> 4, q4 = c & 15;   // q4*4 = first float of 16B chunk
            const float* g = mask + (size_t)(l0 + r) * S_ + s0 + q4 * 4;
            async_load_b128(smask_lds + (unsigned)((r * 64 + q4 * 4) * 4), g);
        }
        // stage K tile (row-major f16)
        for (int c = tid; c < BKV * 16; c += 128) {
            const int r = c >> 4, e4 = c & 15;
            float4 t4 = *(const float4*)(kbase + (size_t)(s0 + r) * (H_ * E_) + e4 * 4);
            *(v4h*)(sK + r * PADH + e4 * 4) = cvt4(t4);
            if (st + 1 < S_ / BKV)   // prefetch next tile's K into cache
                __builtin_prefetch(kbase + (size_t)(s0 + BKV + r) * (H_ * E_) + e4 * 4, 0, 1);
        }
        // stage V tile transposed: sVt[e][key]
        for (int c = tid; c < BKV * 16; c += 128) {
            const int r = c >> 4, e4 = c & 15;
            float4 t4 = *(const float4*)(vbase + (size_t)(s0 + r) * (H_ * E_) + e4 * 4);
            sVt[(e4 * 4 + 0) * PADH + r] = (_Float16)t4.x;
            sVt[(e4 * 4 + 1) * PADH + r] = (_Float16)t4.y;
            sVt[(e4 * 4 + 2) * PADH + r] = (_Float16)t4.z;
            sVt[(e4 * 4 + 3) * PADH + r] = (_Float16)t4.w;
            if (st + 1 < S_ / BKV)   // prefetch next tile's V into cache
                __builtin_prefetch(vbase + (size_t)(s0 + BKV + r) * (H_ * E_) + e4 * 4, 0, 1);
        }
        wait_async0();     // mask tile resident in LDS
        __syncthreads();

        // ---- Kd band: only distance columns this tile pair can touch ----
        const int d0 = l0 - s0;
        int jlo = d0 - (BKV - 1) + MAXP_;
        jlo = jlo < 0 ? 0 : (jlo > 256 ? 256 : jlo);
        int jhi = d0 + (BM - 1) + MAXP_;
        jhi = jhi < 0 ? 0 : (jhi > 256 ? 256 : jhi);
        const int jbase = jlo & ~15;
        const int numt  = (jhi >> 4) - (jbase >> 4) + 1;   // 1..9 column tiles

        const v16h ak0 = load_frag(sK, wave * 16, 0, lane);
        const v16h ak1 = load_frag(sK, wave * 16, 1, lane);
        for (int t = 0; t < numt; ++t) {
            v16h b0 = load_frag(sD, jbase + t * 16, 0, lane);
            v16h b1 = load_frag(sD, jbase + t * 16, 1, lane);
            v8f c = vz;
            c = wmma16(ak0, b0, c);
            c = wmma16(ak1, b1, c);
#pragma unroll
            for (int r = 0; r < 8; ++r)
                sKd[(wave * 16 + r + hi8) * KDPH + t * 16 + nj] = (_Float16)c[r];
        }
        __syncthreads();

        // ---- S_tile = Q x K^T (16 x 64 per wave) ----
        v8f acc[4];
#pragma unroll
        for (int t = 0; t < 4; ++t) {
            v16h b0 = load_frag(sK, t * 16, 0, lane);
            v16h b1 = load_frag(sK, t * 16, 1, lane);
            v8f c = vz;
            c = wmma16(aq0, b0, c);
            c = wmma16(aq1, b1, c);
            acc[t] = c;
        }

        // ---- add relative terms (Toeplitz gather) + mask, scale ----
#pragma unroll
        for (int t = 0; t < 4; ++t) {
            const int s_loc = t * 16 + nj;
#pragma unroll
            for (int r = 0; r < 8; ++r) {
                const int l_loc = wave * 16 + r + hi8;
                int d = (l0 + l_loc) - (s0 + s_loc);
                d = d < -MAXP_ ? -MAXP_ : (d > MAXP_ ? MAXP_ : d);
                const int idx = d + MAXP_;
                const float rel = (float)sQd[l_loc * QDPH + idx] +
                                  (float)sKd[s_loc * KDPH + (idx - jbase)];
                const float mv = sM[l_loc * 64 + s_loc];
                acc[t][r] = (acc[t][r] + rel + mv) * temp;
            }
        }

        // ---- online softmax (row stats live in 16-lane halves) ----
        _Float16* sPw = sP + wave * 16 * PADH;
#pragma unroll
        for (int r = 0; r < 8; ++r) {
            float mx = fmaxf(fmaxf(acc[0][r], acc[1][r]),
                             fmaxf(acc[2][r], acc[3][r]));
            mx = fmaxf(mx, __shfl_xor(mx, 8));
            mx = fmaxf(mx, __shfl_xor(mx, 4));
            mx = fmaxf(mx, __shfl_xor(mx, 2));
            mx = fmaxf(mx, __shfl_xor(mx, 1));
            const float mnew  = fmaxf(m_run[r], mx);
            const float alpha = __expf(m_run[r] - mnew);
            m_run[r] = mnew;
            float psum = 0.f;
#pragma unroll
            for (int t = 0; t < 4; ++t) {
                const float p = __expf(acc[t][r] - mnew);
                acc[t][r] = p;
                psum += p;
            }
            psum += __shfl_xor(psum, 8);
            psum += __shfl_xor(psum, 4);
            psum += __shfl_xor(psum, 2);
            psum += __shfl_xor(psum, 1);
            l_run[r] = l_run[r] * alpha + psum;
#pragma unroll
            for (int t = 0; t < 4; ++t) oacc[t][r] *= alpha;
#pragma unroll
            for (int t = 0; t < 4; ++t)
                sPw[(r + hi8) * PADH + t * 16 + nj] = (_Float16)acc[t][r];
        }
        __syncthreads();

        // ---- O += P x V ----
        const v16h ap0 = load_frag(sPw, 0, 0, lane);
        const v16h ap1 = load_frag(sPw, 0, 1, lane);
#pragma unroll
        for (int t = 0; t < 4; ++t) {
            v16h b0 = load_frag(sVt, t * 16, 0, lane);
            v16h b1 = load_frag(sVt, t * 16, 1, lane);
            oacc[t] = wmma16(ap0, b0, oacc[t]);
            oacc[t] = wmma16(ap1, b1, oacc[t]);
        }
    }

    // ---- epilogue: normalize and store (n,l,h,e) ----
#pragma unroll
    for (int t = 0; t < 4; ++t) {
#pragma unroll
        for (int r = 0; r < 8; ++r) {
            const int l_loc = wave * 16 + r + hi8;
            const int e     = t * 16 + nj;
            out[((size_t)(n * L_ + l0 + l_loc) * H_ + h) * E_ + e] =
                oacc[t][r] / l_run[r];
        }
    }
}

extern "C" void kernel_launch(void* const* d_in, const int* in_sizes, int n_in,
                              void* d_out, int out_size, void* d_ws, size_t ws_size,
                              hipStream_t stream) {
    (void)in_sizes; (void)n_in; (void)out_size; (void)d_ws; (void)ws_size;
    const float* q    = (const float*)d_in[0];
    const float* kk   = (const float*)d_in[1];
    const float* vv   = (const float*)d_in[2];
    const float* mask = (const float*)d_in[3];
    const float* demb = (const float*)d_in[4];
    float* out = (float*)d_out;

    static_assert(SMEM_BYTES <= 160 * 1024, "want 2 blocks per 320KB WGP");
    (void)hipFuncSetAttribute((const void*)FullAttention_relpos_kernel,
                              hipFuncAttributeMaxDynamicSharedMemorySize,
                              (int)SMEM_BYTES);

    dim3 grid(L_ / BM, N_ * H_, 1);
    dim3 block(128, 1, 1);
    FullAttention_relpos_kernel<<<grid, block, SMEM_BYTES, stream>>>(
        q, kk, vv, mask, demb, out);
}
